// singleCausalAttention_60954175864969
// MI455X (gfx1250) — compile-verified, matrix operand order checked by reading, not run on hardware
//
#include <hip/hip_runtime.h>
#include <hip/hip_bf16.h>

typedef __attribute__((ext_vector_type(16))) _Float16     v16h;
typedef __attribute__((ext_vector_type(8)))  float        v8f;
typedef __attribute__((ext_vector_type(4)))  unsigned int v4u;
typedef __attribute__((ext_vector_type(4)))  int          v4i;

#define B_    4
#define T_    4096
#define NIN_  256
#define NOUT_ 64
#define BT_   (B_ * T_)

#define AS1 __attribute__((address_space(1)))
#define AS3 __attribute__((address_space(3)))

#if defined(__gfx1250__) && __has_builtin(__builtin_amdgcn_global_load_async_to_lds_b128) && __has_builtin(__builtin_amdgcn_s_wait_asynccnt)
#define HAVE_ASYNC 1
#else
#define HAVE_ASYNC 0
#endif

__device__ __forceinline__ v8f wmma_f16(v16h a, v16h b, v8f c) {
  // 8 args: (neg_a, A, neg_b, B, c_mod, C, reuse_a, reuse_b)
  return __builtin_amdgcn_wmma_f32_16x16x32_f16(false, a, false, b, (short)0, c,
                                                false, false);
}

// ---------------------------------------------------------------------------
// Kernel 1: QKV projection. y[BT,256] f32 x W[256,64] f32 -> f16 workspace.
//   sel 0: Q [BT][64], pre-scaled by 1/sqrt(64)
//   sel 1: K [BT][64]
//   sel 2: V stored TRANSPOSED as Vt[64][BT]  (so attention can async-stage it)
// Block = 128 threads (4 waves); each wave -> one 16x64 output strip.
// ---------------------------------------------------------------------------
__global__ __launch_bounds__(128) void qkv_proj_kernel(
    const float* __restrict__ y, const float* __restrict__ Wq,
    const float* __restrict__ Wk, const float* __restrict__ Wv,
    _Float16* __restrict__ ws) {
  __shared__ __align__(16) _Float16 Wt[NOUT_][NIN_ + 8];  // W transposed

  const int sel = blockIdx.y;
  const float* W = (sel == 0) ? Wq : (sel == 1) ? Wk : Wv;

  const int tid = threadIdx.x;
  for (int idx = tid; idx < NIN_ * NOUT_; idx += 128) {
    int k = idx >> 6, n = idx & 63;
    Wt[n][k] = (_Float16)W[idx];
  }
  __syncthreads();

  const int wave = tid >> 5, lane = tid & 31;
  const int lr = lane & 15, half = lane >> 4;
  const int row = blockIdx.x * 64 + wave * 16 + lr;
  const float* yrow = y + (size_t)row * NIN_;

  v8f acc[4] = {};

#pragma unroll
  for (int cc = 0; cc < NIN_ / 32; ++cc) {
    const int kb = cc * 32 + half * 8;
    // A fragment (16-bit A layout: lane%16 = row M, K split on lane/16)
    v16h a;
#pragma unroll
    for (int i = 0; i < 8; ++i) {
      a[i]     = (_Float16)yrow[kb + i];
      a[i + 8] = (_Float16)yrow[kb + 16 + i];
    }
#pragma unroll
    for (int t = 0; t < 4; ++t) {
      const _Float16* wr = &Wt[t * 16 + lr][0];
      v16h bf;
#pragma unroll
      for (int i = 0; i < 8; ++i) {
        bf[i]     = wr[kb + i];
        bf[i + 8] = wr[kb + 16 + i];
      }
      acc[t] = wmma_f16(a, bf, acc[t]);
    }
  }

  if (sel == 2) {
    // V transposed: Vt[d][row]; lane writes 8 consecutive rows (16B run)
    _Float16* vt = ws + (size_t)2 * BT_ * NOUT_;
    const int rbase = blockIdx.x * 64 + wave * 16 + half * 8;
#pragma unroll
    for (int t = 0; t < 4; ++t) {
      _Float16* vp = vt + (size_t)(t * 16 + lr) * BT_ + rbase;
#pragma unroll
      for (int r = 0; r < 8; ++r) vp[r] = (_Float16)acc[t][r];
    }
  } else {
    _Float16* out = ws + (size_t)sel * BT_ * NOUT_;
    const float scale = (sel == 0) ? 0.125f : 1.0f;  // 1/sqrt(NOUT) into Q
#pragma unroll
    for (int t = 0; t < 4; ++t)
#pragma unroll
      for (int r = 0; r < 8; ++r) {
        const int rl = r + half * 8;  // C/D layout: VGPR r -> row r (+8 upper)
        out[(size_t)(blockIdx.x * 64 + wave * 16 + rl) * NOUT_ + t * 16 + lr] =
            (_Float16)(acc[t][r] * scale);
      }
  }
}

// ---------------------------------------------------------------------------
// Kernel 2: causal flash attention. f16 Q[BT][64], K[BT][64], Vt[64][BT].
// Block = 128 threads (4 waves) per 64-query tile of one batch.
// K and V tiles are double-buffered in LDS via async global->LDS b128 copies
// (ASYNCcnt), overlapping the next tile's fill with the current tile's WMMAs.
// ---------------------------------------------------------------------------
__global__ __launch_bounds__(128) void flash_attn_kernel(
    const _Float16* __restrict__ ws, float* __restrict__ out) {
  __shared__ __align__(16) _Float16 Kl[2][64][72];   // K tile [key][d]
  __shared__ __align__(16) _Float16 Vl[2][64][72];   // V tile [d][key]
  __shared__ __align__(16) _Float16 Pl[4][16][72];   // per-wave P strip

  const _Float16* Q  = ws;
  const _Float16* K  = ws + (size_t)BT_ * NOUT_;
  const _Float16* Vg = ws + (size_t)2 * BT_ * NOUT_;  // transposed [64][BT]

  const int b = blockIdx.y, qt = blockIdx.x;
  const int q0 = qt * 64;
  const int tid = threadIdx.x, wave = tid >> 5, lane = tid & 31;
  const int lr = lane & 15, half = lane >> 4;

  // Stage one 64x64 K tile + 64x64 V tile: 512 x 16B segments each kind,
  // 4 + 4 issues per thread (8 async ops per wave per tile).
  auto stage = [&](int buf, int jt) {
    const int k0s = jt * 64;
    const _Float16* kb = K + (size_t)(b * T_ + k0s) * NOUT_;
    const _Float16* vb = Vg + (size_t)(b * T_ + k0s);
#pragma unroll
    for (int it = 0; it < 4; ++it) {
      const int idx = tid + it * 128;
      const int rr = idx >> 3, seg = idx & 7;
#if HAVE_ASYNC
      __builtin_amdgcn_global_load_async_to_lds_b128(
          (AS1 v4i*)(kb + (size_t)rr * NOUT_ + seg * 8),
          (AS3 v4i*)&Kl[buf][rr][seg * 8], 0, 0);
      __builtin_amdgcn_global_load_async_to_lds_b128(
          (AS1 v4i*)(vb + (size_t)rr * BT_ + seg * 8),
          (AS3 v4i*)&Vl[buf][rr][seg * 8], 0, 0);
#else
      *(v4u*)&Kl[buf][rr][seg * 8] =
          *(const v4u*)(kb + (size_t)rr * NOUT_ + seg * 8);
      *(v4u*)&Vl[buf][rr][seg * 8] =
          *(const v4u*)(vb + (size_t)rr * BT_ + seg * 8);
#endif
    }
  };

  // Resident Q A-fragments (two 16x32 chunks over d=64; scale pre-folded)
  const _Float16* qrow = Q + (size_t)(b * T_ + q0 + wave * 16 + lr) * NOUT_;
  v16h aq[2];
#pragma unroll
  for (int cc = 0; cc < 2; ++cc) {
    const int kb = cc * 32 + half * 8;
#pragma unroll
    for (int i = 0; i < 8; ++i) {
      aq[cc][i]     = qrow[kb + i];
      aq[cc][i + 8] = qrow[kb + 16 + i];
    }
  }

  float mrun[8], lrun[8];
  v8f Oacc[4] = {};
#pragma unroll
  for (int r = 0; r < 8; ++r) { mrun[r] = -1e30f; lrun[r] = 0.0f; }

#if HAVE_ASYNC
  stage(0, 0);  // prefill buffer 0
#endif

  for (int j = 0; j <= qt; ++j) {
    const int cur = j & 1;
#if HAVE_ASYNC
    if (j < qt) {
      stage(1 - cur, j + 1);                 // overlap next tile with compute
      __builtin_amdgcn_s_wait_asynccnt(8);   // tile j landed (in-order), j+1 inflight
    } else {
      __builtin_amdgcn_s_wait_asynccnt(0);   // drain
    }
#else
    stage(cur, j);
#endif
    __syncthreads();

    // S = Q * K^T  (16x64 strip per wave; B-fragments contiguous in Kl rows)
    v8f sacc[4] = {};
#pragma unroll
    for (int cc = 0; cc < 2; ++cc) {
      const int kb = cc * 32 + half * 8;
#pragma unroll
      for (int t = 0; t < 4; ++t) {
        const _Float16* kr = &Kl[cur][t * 16 + lr][0];
        v16h bf;
#pragma unroll
        for (int i = 0; i < 8; ++i) {
          bf[i]     = kr[kb + i];
          bf[i + 8] = kr[kb + 16 + i];
        }
        sacc[t] = wmma_f16(aq[cc], bf, sacc[t]);
      }
    }

    // Causal mask (only the diagonal tile; k0 == q0 there)
    if (j == qt) {
#pragma unroll
      for (int t = 0; t < 4; ++t)
#pragma unroll
        for (int r = 0; r < 8; ++r) {
          const int qrl = wave * 16 + r + half * 8;
          const int krl = t * 16 + lr;
          if (krl > qrl) sacc[t][r] = -1e30f;
        }
    }

    // Online softmax: row stats butterfly-reduced within each 16-lane half
    float fac[8];
#pragma unroll
    for (int r = 0; r < 8; ++r) {
      float mt = fmaxf(fmaxf(sacc[0][r], sacc[1][r]),
                       fmaxf(sacc[2][r], sacc[3][r]));
#pragma unroll
      for (int m = 1; m <= 8; m <<= 1) mt = fmaxf(mt, __shfl_xor(mt, m, 32));
      const float mn = fmaxf(mrun[r], mt);
      fac[r] = __expf(mrun[r] - mn);
      mrun[r] = mn;
    }
#pragma unroll
    for (int t = 0; t < 4; ++t)
#pragma unroll
      for (int r = 0; r < 8; ++r) sacc[t][r] = __expf(sacc[t][r] - mrun[r]);
#pragma unroll
    for (int r = 0; r < 8; ++r) {
      float rs = sacc[0][r] + sacc[1][r] + sacc[2][r] + sacc[3][r];
#pragma unroll
      for (int m = 1; m <= 8; m <<= 1) rs += __shfl_xor(rs, m, 32);
      lrun[r] = lrun[r] * fac[r] + rs;
#pragma unroll
      for (int t = 0; t < 4; ++t) Oacc[t][r] *= fac[r];
    }

    // P: C-layout -> wave-private LDS strip -> A-layout fragments
#pragma unroll
    for (int t = 0; t < 4; ++t)
#pragma unroll
      for (int r = 0; r < 8; ++r)
        Pl[wave][r + half * 8][t * 16 + lr] = (_Float16)sacc[t][r];

#pragma unroll
    for (int cc = 0; cc < 2; ++cc) {
      const int kb = cc * 32 + half * 8;
      v16h ap;
      const _Float16* pr = &Pl[wave][lr][0];
#pragma unroll
      for (int i = 0; i < 8; ++i) {
        ap[i]     = pr[kb + i];
        ap[i + 8] = pr[kb + 16 + i];
      }
#pragma unroll
      for (int t = 0; t < 4; ++t) {
        const _Float16* vr = &Vl[cur][t * 16 + lr][0];
        v16h bf;
#pragma unroll
        for (int i = 0; i < 8; ++i) {
          bf[i]     = vr[kb + i];
          bf[i + 8] = vr[kb + 16 + i];
        }
        Oacc[t] = wmma_f16(ap, bf, Oacc[t]);
      }
    }
    __syncthreads();  // all reads of buf `cur` done before it is re-staged
  }

  // Epilogue: O / l, fp32 store
#pragma unroll
  for (int t = 0; t < 4; ++t)
#pragma unroll
    for (int r = 0; r < 8; ++r) {
      const int rl = wave * 16 + r + half * 8;
      out[((size_t)(b * T_) + q0 + rl) * NOUT_ + t * 16 + lr] =
          Oacc[t][r] / lrun[r];
    }
}

extern "C" void kernel_launch(void* const* d_in, const int* in_sizes, int n_in,
                              void* d_out, int out_size, void* d_ws, size_t ws_size,
                              hipStream_t stream) {
  const float* y  = (const float*)d_in[0];
  const float* Wq = (const float*)d_in[1];
  const float* Wk = (const float*)d_in[2];
  const float* Wv = (const float*)d_in[3];
  float* outp = (float*)d_out;
  _Float16* ws = (_Float16*)d_ws;  // Q[BT][64] | K[BT][64] | Vt[64][BT], 6 MB

  qkv_proj_kernel<<<dim3(BT_ / 64, 3), 128, 0, stream>>>(y, Wq, Wk, Wv, ws);
  flash_attn_kernel<<<dim3(T_ / 64, B_), 128, 0, stream>>>(ws, outp);
}
